// SAT_3770981286776
// MI455X (gfx1250) — matrix operand order repaired and checked by reading, not compiled, not run on hardware
//
#include <hip/hip_runtime.h>
#include <hip/hip_bf16.h>
#include <math.h>

typedef _Float16 f16;
typedef __attribute__((ext_vector_type(16))) _Float16 v16h;
typedef __attribute__((ext_vector_type(8)))  _Float16 v8h;
typedef __attribute__((ext_vector_type(8)))  float    v8f;

// Problem constants
static constexpr int kB   = 64;
static constexpr int kNT  = 198;      // 1 cls + 196 patches + 1 loc
static constexpr int kC   = 768;
static constexpr int kNH  = 12;
static constexpr int kHD  = 64;
static constexpr int kCQ  = 3 * kC;   // 2304
static constexpr int kCH  = 4 * kC;   // 3072
static constexpr int kM   = kB * kNT; // 12672 = 198*64
static constexpr int kMP  = kB * 196; // 12544 = 196*64
static constexpr float kSCALE = 0.125f; // 64^-0.5

// ---------------------------------------------------------------------------
// WMMA GEMM:  C[M,N] = A[M,K] * W[N,K]^T  (+bias), f16 in / f32 out.
// Requires M%64==0, N%64==0, K%64==0. Block = 128 threads = 4 waves,
// block tile 64x64, each wave owns a 32x32 tile (2x2 wmma_f32_16x16x32_f16,
// two K-slices per stage). Double-buffered LDS fed by CDNA5 async-to-LDS
// loads (GLOBAL_LOAD_ASYNC_TO_LDS_B128, tracked by ASYNCcnt).
// ---------------------------------------------------------------------------
#define LROWH 72   // LDS row pitch in halves (144 B, 16B-aligned, padded)

__device__ __forceinline__ void async_b128(unsigned lds_addr, const f16* gaddr) {
  asm volatile("global_load_async_to_lds_b128 %0, %1, off"
               :: "v"(lds_addr), "v"(gaddr)
               : "memory");
}
__device__ __forceinline__ void wait_async0() {
  asm volatile("s_wait_asynccnt 0x0" ::: "memory");
}

__global__ __launch_bounds__(128) void k_gemm(const f16* __restrict__ A,
                                              const f16* __restrict__ W,
                                              float* __restrict__ Cout,
                                              const float* __restrict__ bias,
                                              int M, int N, int K) {
  __shared__ f16 As[2][64][LROWH];
  __shared__ f16 Ws[2][64][LROWH];
  const int tid  = threadIdx.x;
  const int lane = tid & 31;
  const int wid  = tid >> 5;   // 0..3
  const int wm   = wid >> 1;   // 0..1
  const int wn   = wid & 1;    // 0..1
  const int bm   = blockIdx.y * 64;
  const int bn   = blockIdx.x * 64;

  v8f acc[2][2];
  acc[0][0] = {}; acc[0][1] = {}; acc[1][0] = {}; acc[1][1] = {};

  // Staging map: 128 threads x 32 halves (4 x b128) per matrix per stage.
  const int lrow = tid >> 1;         // 0..63
  const int lseg = (tid & 1) * 32;   // 0 or 32 (halves)

  const f16* agp = A + (size_t)(bm + lrow) * K + lseg;
  const f16* wgp = W + (size_t)(bn + lrow) * K + lseg;
  const unsigned laA0 = (unsigned)(size_t)(&As[0][lrow][lseg]);
  const unsigned laW0 = (unsigned)(size_t)(&Ws[0][lrow][lseg]);
  const unsigned laA1 = (unsigned)(size_t)(&As[1][lrow][lseg]);
  const unsigned laW1 = (unsigned)(size_t)(&Ws[1][lrow][lseg]);

  // Prologue: stage 0
#pragma unroll
  for (int c2 = 0; c2 < 4; ++c2) {
    async_b128(laA0 + c2 * 16, agp + c2 * 8);
    async_b128(laW0 + c2 * 16, wgp + c2 * 8);
  }
  wait_async0();
  __syncthreads();

  int stage = 0;
  for (int k0 = 0; k0 < K; k0 += 64) {
    // Kick off next stage while computing this one.
    if (k0 + 64 < K) {
      const unsigned la = stage ? laA0 : laA1;
      const unsigned lw = stage ? laW0 : laW1;
      const f16* an = agp + k0 + 64;
      const f16* wn_ = wgp + k0 + 64;
#pragma unroll
      for (int c2 = 0; c2 < 4; ++c2) {
        async_b128(la + c2 * 16, an + c2 * 8);
        async_b128(lw + c2 * 16, wn_ + c2 * 8);
      }
      if (k0 + 128 < K) {  // warm L2 for the stage after next
        __builtin_prefetch(agp + k0 + 128, 0, 1);
        __builtin_prefetch(wgp + k0 + 128, 0, 1);
      }
    }

    const int half = lane >> 4;   // 0/1: lane-half K selector
#pragma unroll
    for (int kk = 0; kk < 2; ++kk) {   // two 32-wide K slices per stage
      // B fragments (32x16): lane holds col N=lane%16; K = {half*8..+8} U {16+half*8..+8}
      v16h bfrag[2];
#pragma unroll
      for (int j = 0; j < 2; ++j) {
        const int wrow = wn * 32 + j * 16 + (lane & 15);
        v8h b0 = *(const v8h*)(&Ws[stage][wrow][kk * 32 + half * 8]);
        v8h b1 = *(const v8h*)(&Ws[stage][wrow][kk * 32 + 16 + half * 8]);
#pragma unroll
        for (int e = 0; e < 8; ++e) { bfrag[j][e] = b0[e]; bfrag[j][e + 8] = b1[e]; }
      }
#pragma unroll
      for (int i = 0; i < 2; ++i) {
        // A fragment (16x32): lane row M=lane%16; K = {half*8..+8} U {16+half*8..+8}
        const int arow = wm * 32 + i * 16 + (lane & 15);
        v8h a0 = *(const v8h*)(&As[stage][arow][kk * 32 + half * 8]);
        v8h a1 = *(const v8h*)(&As[stage][arow][kk * 32 + 16 + half * 8]);
        v16h av;
#pragma unroll
        for (int e = 0; e < 8; ++e) { av[e] = a0[e]; av[e + 8] = a1[e]; }
        acc[i][0] = __builtin_amdgcn_wmma_f32_16x16x32_f16(
            false, av, false, bfrag[0], (short)0, acc[i][0], false, false);
        acc[i][1] = __builtin_amdgcn_wmma_f32_16x16x32_f16(
            false, av, false, bfrag[1], (short)0, acc[i][1], false, false);
      }
    }
    wait_async0();     // next stage landed in LDS (this wave's ASYNCcnt)
    __syncthreads();   // all waves done reading current stage
    stage ^= 1;
  }

  // D layout: lane%16 = N col; VGPR r -> M = r + (lane>=16 ? 8 : 0)
#pragma unroll
  for (int i = 0; i < 2; ++i) {
#pragma unroll
    for (int j = 0; j < 2; ++j) {
      const int col   = bn + wn * 32 + j * 16 + (lane & 15);
      const int rbase = bm + wm * 32 + i * 16 + ((lane >> 4) ? 8 : 0);
      const float bv  = bias ? bias[col] : 0.f;
#pragma unroll
      for (int r = 0; r < 8; ++r)
        Cout[(size_t)(rbase + r) * N + col] = acc[i][j][r] + bv;
    }
  }
}

// ---------------------------------------------------------------------------
// Elementwise / small kernels
// ---------------------------------------------------------------------------
__global__ void k_f32_to_f16(const float* __restrict__ s, f16* __restrict__ d,
                             long long n) {
  for (long long i = (long long)blockIdx.x * blockDim.x + threadIdx.x; i < n;
       i += (long long)gridDim.x * blockDim.x)
    d[i] = (f16)s[i];
}

__global__ __launch_bounds__(256) void k_im2col(const float* __restrict__ x,
                                                f16* __restrict__ out) {
  const int bp = blockIdx.x;          // b*196 + p
  const int p  = bp % 196, b = bp / 196;
  const int ph = p / 14, pw = p % 14;
  for (int kc = threadIdx.x; kc < kC; kc += 256) {
    const int c = kc >> 8, ij = kc & 255, i = ij >> 4, j = ij & 15;
    out[(size_t)bp * kC + kc] =
        (f16)x[((size_t)(b * 3 + c) * 224 + ph * 16 + i) * 224 + pw * 16 + j];
  }
}

__global__ __launch_bounds__(256) void k_assemble(
    const float* __restrict__ pemb, const float* __restrict__ patch_b,
    const float* __restrict__ cls, const float* __restrict__ loc,
    const float* __restrict__ pos, const float* __restrict__ loce,
    float* __restrict__ t) {
  const int idx = blockIdx.x;
  const int n = idx % kNT, b = idx / kNT;
  for (int c = threadIdx.x; c < kC; c += 256) {
    float v, pe;
    if (n == 0)            { v = cls[c];  pe = pos[c]; }
    else if (n == kNT - 1) { v = loc[c];  pe = loce[c]; }
    else { v = pemb[((size_t)(b * 196 + n - 1)) * kC + c] + patch_b[c];
           pe = pos[(size_t)n * kC + c]; }
    t[((size_t)(b * kNT + n)) * kC + c] = v + pe;
  }
}

__global__ __launch_bounds__(256) void k_layernorm(
    const float* __restrict__ x, const float* __restrict__ w,
    const float* __restrict__ b, f16* __restrict__ out16,
    float* __restrict__ out32, int ncol) {
  __shared__ float red[256];
  const int row = blockIdx.x;
  const float* xr = x + (size_t)row * ncol;
  float s = 0.f;
  for (int c = threadIdx.x; c < ncol; c += 256) s += xr[c];
  red[threadIdx.x] = s; __syncthreads();
  for (int st = 128; st > 0; st >>= 1) {
    if (threadIdx.x < st) red[threadIdx.x] += red[threadIdx.x + st];
    __syncthreads();
  }
  const float mean = red[0] / ncol;
  __syncthreads();
  float v = 0.f;
  for (int c = threadIdx.x; c < ncol; c += 256) {
    const float d = xr[c] - mean; v += d * d;
  }
  red[threadIdx.x] = v; __syncthreads();
  for (int st = 128; st > 0; st >>= 1) {
    if (threadIdx.x < st) red[threadIdx.x] += red[threadIdx.x + st];
    __syncthreads();
  }
  const float rstd = rsqrtf(red[0] / ncol + 1e-5f);
  for (int c = threadIdx.x; c < ncol; c += 256) {
    const float y = (xr[c] - mean) * rstd * w[c] + b[c];
    if (out16) out16[(size_t)row * ncol + c] = (f16)y;
    if (out32) out32[(size_t)row * ncol + c] = y;
  }
}

__global__ __launch_bounds__(256) void k_scores(const float* __restrict__ qkv,
                                                float* __restrict__ attn) {
  int idx = blockIdx.x;                 // (b*NH + h)*NT + nq
  const int nq = idx % kNT; idx /= kNT;
  const int h = idx % kNH, b = idx / kNH;
  __shared__ float qs[kHD];
  const float* qrow = qkv + ((size_t)(b * kNT + nq)) * kCQ + h * kHD;
  if (threadIdx.x < kHD) qs[threadIdx.x] = qrow[threadIdx.x];
  __syncthreads();
  for (int k = threadIdx.x; k < kNT; k += 256) {
    const float* krow = qkv + ((size_t)(b * kNT + k)) * kCQ + kC + h * kHD;
    float s = 0.f;
    for (int d0 = 0; d0 < kHD; ++d0) s += qs[d0] * krow[d0];
    attn[(((size_t)(b * kNH + h)) * kNT + nq) * kNT + k] = s * kSCALE;
  }
}

__global__ __launch_bounds__(256) void k_mask(const float* __restrict__ attn,
                                              float* __restrict__ mask) {
  const int b = blockIdx.x;
  for (int k = threadIdx.x; k < kNT; k += 256) {
    float s = 0.f;
    for (int h = 0; h < kNH; ++h)
      s += attn[(((size_t)(b * kNH + h)) * kNT + (kNT - 1)) * kNT + k];
    s *= (1.f / kNH);
    mask[(size_t)b * kNT + k] = 1.f / (1.f + expf(-s));
  }
}

__global__ __launch_bounds__(256) void k_softmax_av(
    const float* __restrict__ attn, const float* __restrict__ qkv,
    const float* __restrict__ mask, f16* __restrict__ o16) {
  int idx = blockIdx.x;
  const int nq = idx % kNT; idx /= kNT;
  const int h = idx % kNH, b = idx / kNH;
  __shared__ float p[kNT];
  __shared__ float red[256];
  const float* arow = attn + (((size_t)(b * kNH + h)) * kNT + nq) * kNT;
  float m = -1e30f;
  for (int k = threadIdx.x; k < kNT; k += 256) m = fmaxf(m, arow[k]);
  red[threadIdx.x] = m; __syncthreads();
  for (int st = 128; st > 0; st >>= 1) {
    if (threadIdx.x < st)
      red[threadIdx.x] = fmaxf(red[threadIdx.x], red[threadIdx.x + st]);
    __syncthreads();
  }
  m = red[0]; __syncthreads();
  float s = 0.f;
  for (int k = threadIdx.x; k < kNT; k += 256) {
    const float e = expf(arow[k] - m); p[k] = e; s += e;
  }
  red[threadIdx.x] = s; __syncthreads();
  for (int st = 128; st > 0; st >>= 1) {
    if (threadIdx.x < st) red[threadIdx.x] += red[threadIdx.x + st];
    __syncthreads();
  }
  const float inv = 1.f / red[0];
  __syncthreads();
  for (int k = threadIdx.x; k < kNT; k += 256) {
    float pv = p[k] * inv;
    if (mask) pv *= mask[(size_t)b * kNT + k];
    p[k] = pv;
  }
  __syncthreads();
  for (int d0 = threadIdx.x; d0 < kHD; d0 += 256) {
    float acc = 0.f;
    for (int k = 0; k < kNT; ++k)
      acc += p[k] * qkv[((size_t)(b * kNT + k)) * kCQ + 2 * kC + h * kHD + d0];
    o16[((size_t)(b * kNT + nq)) * kC + h * kHD + d0] = (f16)acc;
  }
}

__global__ void k_bias_residual(float* __restrict__ t,
                                const float* __restrict__ g,
                                const float* __restrict__ bias, long long n,
                                int ncol) {
  for (long long i = (long long)blockIdx.x * blockDim.x + threadIdx.x; i < n;
       i += (long long)gridDim.x * blockDim.x)
    t[i] += g[i] + bias[(int)(i % ncol)];
}

__global__ void k_gelu(const float* __restrict__ g,
                       const float* __restrict__ bias, f16* __restrict__ out,
                       long long n, int ncol) {
  for (long long i = (long long)blockIdx.x * blockDim.x + threadIdx.x; i < n;
       i += (long long)gridDim.x * blockDim.x) {
    const float x = g[i] + bias[(int)(i % ncol)];
    out[i] = (f16)(0.5f * x * (1.f + erff(x * 0.70710678118f)));
  }
}

// Head: S[b,c,di,dj] = clipped spatial sums -> logits via tiny contraction.
__global__ __launch_bounds__(256) void k_headsum(const float* __restrict__ nrm,
                                                 float* __restrict__ S) {
  const int idx = blockIdx.x;             // b*kC + c
  const int c = idx % kC, b = idx / kC;
  __shared__ float s9[9];
  if (threadIdx.x < 9) s9[threadIdx.x] = 0.f;
  __syncthreads();
  const int p = threadIdx.x;
  if (p < 196) {
    const int r = p / 14, cl = p % 14;
    const float v = nrm[((size_t)(b * kNT + 1 + p)) * kC + c];
    for (int di = 0; di < 3; ++di) {
      const bool rok = (di == 0) ? (r <= 12) : (di == 1) ? true : (r >= 1);
      if (!rok) continue;
      for (int dj = 0; dj < 3; ++dj) {
        const bool cok = (dj == 0) ? (cl <= 12) : (dj == 1) ? true : (cl >= 1);
        if (cok) atomicAdd(&s9[di * 3 + dj], v);
      }
    }
  }
  __syncthreads();
  if (threadIdx.x < 9)
    S[((size_t)(b * kC + c)) * 9 + threadIdx.x] = s9[threadIdx.x];
}

__global__ __launch_bounds__(256) void k_headlogits(
    const float* __restrict__ S, const float* __restrict__ hw,
    const float* __restrict__ hb, float* __restrict__ out) {
  const int idx = blockIdx.x;           // b*200 + o
  const int o = idx % 200, b = idx / 200;
  __shared__ float red[256];
  float acc = 0.f;
  const float* Sb = S + (size_t)b * kC * 9;
  const float* wo = hw + (size_t)o * kC * 9;
  for (int i = threadIdx.x; i < kC * 9; i += 256) acc += Sb[i] * wo[i];
  red[threadIdx.x] = acc; __syncthreads();
  for (int st = 128; st > 0; st >>= 1) {
    if (threadIdx.x < st) red[threadIdx.x] += red[threadIdx.x + st];
    __syncthreads();
  }
  if (threadIdx.x == 0) out[(size_t)b * 200 + o] = red[0] / 196.f + hb[o];
}

__global__ __launch_bounds__(256) void k_maskall(const float* __restrict__ m3,
                                                 float* __restrict__ mall,
                                                 float* __restrict__ omean) {
  const int b = blockIdx.x;
  __shared__ float red[256];
  float s = 0.f;
  for (int p = threadIdx.x; p < 196; p += 256) {
    const float m = (m3[((size_t)(0 * kB + b)) * kNT + p + 1] +
                     m3[((size_t)(1 * kB + b)) * kNT + p + 1] +
                     m3[((size_t)(2 * kB + b)) * kNT + p + 1]) * (1.f / 3.f);
    mall[(size_t)b * 196 + p] = m;
    s += m;
  }
  red[threadIdx.x] = s; __syncthreads();
  for (int st = 128; st > 0; st >>= 1) {
    if (threadIdx.x < st) red[threadIdx.x] += red[threadIdx.x + st];
    __syncthreads();
  }
  if (threadIdx.x == 0) omean[b] = red[0] / 196.f;
}

__global__ __launch_bounds__(256) void k_maskreg(const float* __restrict__ mall,
                                                 const float* __restrict__ gk,
                                                 float* __restrict__ oreg) {
  const int b = blockIdx.x;
  __shared__ float red[256];
  float s = 0.f;
  for (int p = threadIdx.x; p < 196; p += 256) {
    const int r = p / 14, cl = p % 14;
    float acc = 0.f;
    for (int di = 0; di < 3; ++di)
      for (int dj = 0; dj < 3; ++dj) {
        const int rr = r + di - 1, cc = cl + dj - 1;
        if (rr >= 0 && rr < 14 && cc >= 0 && cc < 14)
          acc += gk[di * 3 + dj] * mall[(size_t)b * 196 + rr * 14 + cc];
      }
    s += (1.f - acc) * acc;
  }
  red[threadIdx.x] = s; __syncthreads();
  for (int st = 128; st > 0; st >>= 1) {
    if (threadIdx.x < st) red[threadIdx.x] += red[threadIdx.x + st];
    __syncthreads();
  }
  if (threadIdx.x == 0) oreg[b] = red[0] / 196.f;
}

// ---------------------------------------------------------------------------
extern "C" void kernel_launch(void* const* d_in, const int* in_sizes, int n_in,
                              void* d_out, int out_size, void* d_ws,
                              size_t ws_size, hipStream_t stream) {
  const float* x       = (const float*)d_in[0];
  const float* patch_w = (const float*)d_in[1];
  const float* patch_b = (const float*)d_in[2];
  const float* cls_tok = (const float*)d_in[3];
  const float* loc_tok = (const float*)d_in[4];
  const float* pos_emb = (const float*)d_in[5];
  const float* loc_emb = (const float*)d_in[6];
  const float* ln1_w   = (const float*)d_in[7];
  const float* ln1_b   = (const float*)d_in[8];
  const float* qkv_w   = (const float*)d_in[9];
  const float* proj_w  = (const float*)d_in[10];
  const float* proj_b  = (const float*)d_in[11];
  const float* ln2_w   = (const float*)d_in[12];
  const float* ln2_b   = (const float*)d_in[13];
  const float* fc1_w   = (const float*)d_in[14];
  const float* fc1_b   = (const float*)d_in[15];
  const float* fc2_w   = (const float*)d_in[16];
  const float* fc2_b   = (const float*)d_in[17];
  const float* norm_w  = (const float*)d_in[18];
  const float* norm_b  = (const float*)d_in[19];
  const float* head_w  = (const float*)d_in[20];
  const float* head_b  = (const float*)d_in[21];
  const float* gk      = (const float*)d_in[22];

  char* ws = (char*)d_ws;
  size_t off = 0;
  auto alloc = [&](size_t bytes) -> void* {
    void* p = ws + off;
    off = (off + bytes + 255) & ~(size_t)255;
    return p;
  };

  float* t     = (float*)alloc((size_t)kM * kC * 4);
  float* tmp   = (float*)alloc((size_t)kM * kCH * 4);        // biggest GEMM out
  f16*   act16 = (f16*)alloc((size_t)kM * kCH * 2);          // GEMM A operand
  f16*   w16   = (f16*)alloc((size_t)kCH * kC * 2);          // f16 weights
  float* attn  = (float*)alloc((size_t)kB * kNH * kNT * kNT * 4);
  float* mask3 = (float*)alloc((size_t)3 * kB * kNT * 4);
  float* mall  = (float*)alloc((size_t)kB * 196 * 4);

  float* out = (float*)d_out;  // [B*200 logits | B mask-mean | B reg]

  // --- Patch embedding as WMMA GEMM (im2col is free: stride==patch) ---
  k_im2col<<<kMP, 256, 0, stream>>>(x, act16);
  k_f32_to_f16<<<2048, 256, 0, stream>>>(patch_w, w16, (long long)kC * kC);
  k_gemm<<<dim3(kC / 64, kMP / 64), 128, 0, stream>>>(act16, w16, tmp, nullptr,
                                                      kMP, kC, kC);
  k_assemble<<<kB * kNT, 256, 0, stream>>>(tmp, patch_b, cls_tok, loc_tok,
                                           pos_emb, loc_emb, t);

  // --- Transformer layers ---
  for (int d = 0; d < 12; ++d) {
    k_layernorm<<<kM, 256, 0, stream>>>(t, ln1_w + (size_t)d * kC,
                                        ln1_b + (size_t)d * kC, act16, nullptr,
                                        kC);
    k_f32_to_f16<<<2048, 256, 0, stream>>>(qkv_w + (size_t)d * kCQ * kC, w16,
                                           (long long)kCQ * kC);
    k_gemm<<<dim3(kCQ / 64, kM / 64), 128, 0, stream>>>(act16, w16, tmp,
                                                        nullptr, kM, kCQ, kC);
    k_scores<<<kB * kNH * kNT, 256, 0, stream>>>(tmp, attn);
    const float* mptr = nullptr;
    if (d >= 9) {
      float* mslice = mask3 + (size_t)(d - 9) * kB * kNT;
      k_mask<<<kB, 256, 0, stream>>>(attn, mslice);
      mptr = mslice;
    }
    k_softmax_av<<<kB * kNH * kNT, 256, 0, stream>>>(attn, tmp, mptr, act16);

    k_f32_to_f16<<<2048, 256, 0, stream>>>(proj_w + (size_t)d * kC * kC, w16,
                                           (long long)kC * kC);
    k_gemm<<<dim3(kC / 64, kM / 64), 128, 0, stream>>>(act16, w16, tmp, nullptr,
                                                       kM, kC, kC);
    k_bias_residual<<<4096, 256, 0, stream>>>(t, tmp, proj_b + (size_t)d * kC,
                                              (long long)kM * kC, kC);

    k_layernorm<<<kM, 256, 0, stream>>>(t, ln2_w + (size_t)d * kC,
                                        ln2_b + (size_t)d * kC, act16, nullptr,
                                        kC);
    k_f32_to_f16<<<2048, 256, 0, stream>>>(fc1_w + (size_t)d * kCH * kC, w16,
                                           (long long)kCH * kC);
    k_gemm<<<dim3(kCH / 64, kM / 64), 128, 0, stream>>>(act16, w16, tmp,
                                                        nullptr, kM, kCH, kC);
    k_gelu<<<4096, 256, 0, stream>>>(tmp, fc1_b + (size_t)d * kCH, act16,
                                     (long long)kM * kCH, kCH);
    k_f32_to_f16<<<2048, 256, 0, stream>>>(fc2_w + (size_t)d * kC * kCH, w16,
                                           (long long)kC * kCH);
    k_gemm<<<dim3(kC / 64, kM / 64), 128, 0, stream>>>(act16, w16, tmp, nullptr,
                                                       kM, kC, kCH);
    k_bias_residual<<<4096, 256, 0, stream>>>(t, tmp, fc2_b + (size_t)d * kC,
                                              (long long)kM * kC, kC);
  }

  // --- Final LN + CAM head (shifted-sum trick) + mask outputs ---
  float* nrm = (float*)act16;  // reuse scratch as f32 buffer
  k_layernorm<<<kM, 256, 0, stream>>>(t, norm_w, norm_b, nullptr, nrm, kC);
  float* S = (float*)w16;      // B*768*9 floats fit in weight scratch
  k_headsum<<<kB * kC, 256, 0, stream>>>(nrm, S);
  k_headlogits<<<kB * 200, 256, 0, stream>>>(S, head_w, head_b, out);
  k_maskall<<<kB, 256, 0, stream>>>(mask3, mall, out + (size_t)kB * 200);
  k_maskreg<<<kB, 256, 0, stream>>>(mall, gk, out + (size_t)kB * 200 + kB);
}
